// SemanticAugSingleHeadAttnModule_80779744903218
// MI455X (gfx1250) — matrix operand order repaired and checked by reading, not compiled, not run on hardware
//
#include <hip/hip_runtime.h>
#include <hip/hip_bf16.h>

#define DIM   128
#define BATCH 4
#define NQ    4096
#define NK    4096
#define SCALE 0.08838834764831845f  // 128^-0.5

typedef __attribute__((ext_vector_type(16))) _Float16 v16h;
typedef __attribute__((ext_vector_type(8)))  float    v8f;

// A-fragment (16x32 f16, MxK): lane (m + 16*hl) element pair 2v..2v+1 holds
// K = 16*hl + 2*v + {0,1}   (contiguous order, matches B fragments)
__device__ __forceinline__ int a_koff(int v, int hl) {
  return (hl << 4) + (v << 1);
}

static __device__ __forceinline__ v8f wmma_f16(v16h a, v16h b, v8f c) {
  return __builtin_amdgcn_wmma_f32_16x16x32_f16(false, a, false, b, (short)0, c,
                                                false, false);
}

// ---------------------------------------------------------------------------
// Generic C = A[rows,128] @ W[128,128] + bias, via WMMA f16, f32 accumulate.
// OutT = _Float16 (projection to ws) or float (final output).
// TRANS: store output transposed per batch: Cout[b][n][r]  (used for V).
// ---------------------------------------------------------------------------
template <typename OutT, bool TRANS>
__global__ __launch_bounds__(256) void gemm128_wmma(
    const float* __restrict__ A, const float* __restrict__ W,
    const float* __restrict__ bias, OutT* __restrict__ Cout, int total_rows,
    int batch_rows) {
  __shared__ __align__(32) _Float16 Wt[128][128];  // Wt[n][k] = W[k][n] (f16)
  const int t = threadIdx.x;
  for (int i = t; i < 128 * 128; i += 256) {
    int k = i >> 7, n = i & 127;
    Wt[n][k] = (_Float16)W[i];
  }
  __syncthreads();

  const int lane = t & 31, wid = t >> 5;
  const int hl = lane >> 4, nlo = lane & 15;
  const int row0 = blockIdx.x * 128 + wid * 16;
  if (row0 >= total_rows) return;

  // Load 4 A-fragments (this wave's 16 rows, K = 0..127)
  v16h af[4];
  {
    const float* Ar = A + (size_t)(row0 + nlo) * DIM;
    #pragma unroll
    for (int c = 0; c < 4; ++c) {
      v16h tmp;
      #pragma unroll
      for (int v = 0; v < 8; ++v) {
        int k0 = c * 32 + a_koff(v, hl);
        tmp[2 * v]     = (_Float16)Ar[k0];
        tmp[2 * v + 1] = (_Float16)Ar[k0 + 1];
      }
      af[c] = tmp;
    }
  }

  for (int nt = 0; nt < 8; ++nt) {
    const int n = nt * 16 + nlo;
    const float bv = bias ? bias[n] : 0.0f;
    v8f acc;
    #pragma unroll
    for (int i = 0; i < 8; ++i) acc[i] = bv;
    // B-frag lane data = 32 contiguous bytes: Wt[n][c*32 + 16*hl .. +16)
    const v16h* Wrow = (const v16h*)&Wt[n][0];
    #pragma unroll
    for (int c = 0; c < 4; ++c)
      acc = wmma_f16(af[c], Wrow[2 * c + hl], acc);
    #pragma unroll
    for (int v = 0; v < 8; ++v) {
      const int mrow = row0 + v + 8 * hl;  // C layout: m = v + 8*hl
      const float val = acc[v];
      if (TRANS) {
        int b = mrow / batch_rows, r = mrow - b * batch_rows;
        Cout[((size_t)b * DIM + n) * batch_rows + r] = (OutT)val;
      } else {
        Cout[(size_t)mrow * DIM + n] = (OutT)val;
      }
    }
  }
}

// ---------------------------------------------------------------------------
// kb[b][k] = value[b][k] . vs_w + vs_b + att_bias[b][k]
// ---------------------------------------------------------------------------
__global__ __launch_bounds__(256) void kbias_kernel(
    const float* __restrict__ value, const float* __restrict__ att_bias,
    const float* __restrict__ vs_w, const float* __restrict__ vs_b,
    float* __restrict__ kb) {
  int i = blockIdx.x * 256 + threadIdx.x;
  if (i >= BATCH * NK) return;
  const float* vr = value + (size_t)i * DIM;
  float s = vs_b[0];
  #pragma unroll 8
  for (int d = 0; d < DIM; ++d) s += vr[d] * vs_w[d];
  kb[i] = s + att_bias[i];
}

// ---------------------------------------------------------------------------
// Flash attention: 8 waves/WG, 16 q-rows/wave, 32-key tiles, online softmax.
// K and V tiles staged ONCE per workgroup in LDS (8x cut in L2 traffic vs
// per-wave loads), fragments read as contiguous 32B v16h from LDS.
// Qp: f16 [B][NQ][128]  Kp: f16 [B][NK][128]  Vt: f16 [B][128][NK]
// kb: f32 [B][NK] (semantic + att_bias)       O:  f32 [B][NQ][128]
// ---------------------------------------------------------------------------
__global__ __launch_bounds__(256) void flash_attn_wmma(
    const _Float16* __restrict__ Qp, const _Float16* __restrict__ Kp,
    const _Float16* __restrict__ Vt, const float* __restrict__ kb,
    float* __restrict__ O) {
  const int qblocks = NQ / 128;
  const int b = blockIdx.x / qblocks;
  const int t = threadIdx.x;
  const int wid = t >> 5;
  const int lane = t & 31;
  const int hl = lane >> 4, nlo = lane & 15;
  const int qrow0 = (blockIdx.x % qblocks) * 128 + wid * 16;

  const _Float16* Qb = Qp + ((size_t)b * NQ + qrow0) * DIM;
  const _Float16* Kb = Kp + (size_t)b * NK * DIM;
  const _Float16* Vb = Vt + (size_t)b * DIM * NK;
  const float* kbb = kb + (size_t)b * NK;

  __shared__ __align__(32) _Float16 Kl[32][128];       // 8KB, key-major
  __shared__ __align__(32) _Float16 Vl[128][32];       // 8KB, d-major
  __shared__ __align__(32) _Float16 pbuf[8][16][32];   // 8KB, per-wave P

  // Q fragments (16 rows x 128, four 16x32 A-frags), resident for whole loop
  v16h qf[4];
  {
    const _Float16* Qr = Qb + (size_t)nlo * DIM;
    #pragma unroll
    for (int c = 0; c < 4; ++c) {
      v16h tmp;
      #pragma unroll
      for (int v = 0; v < 8; ++v) {
        int k0 = c * 32 + a_koff(v, hl);
        tmp[2 * v]     = Qr[k0];
        tmp[2 * v + 1] = Qr[k0 + 1];
      }
      qf[c] = tmp;
    }
  }

  v8f of[8];  // O accumulators, 16 rows x 128 (eight 16x16 f32 C-frags)
  #pragma unroll
  for (int c = 0; c < 8; ++c)
    #pragma unroll
    for (int i = 0; i < 8; ++i) of[c][i] = 0.0f;

  float rm[8], rs[8];  // per-row running max / sum (row m = v + 8*hl)
  #pragma unroll
  for (int v = 0; v < 8; ++v) { rm[v] = -1e30f; rs[v] = 0.0f; }

  for (int kt = 0; kt < NK / 32; ++kt) {
    const int key0 = kt * 32;

    // ---- cooperative staging: K tile (contiguous 8KB) + V tile ----
    __syncthreads();  // previous iteration's Kl/Vl reads complete
    {
      const v16h* srcK = (const v16h*)(Kb + (size_t)key0 * DIM);
      ((v16h*)Kl)[t] = srcK[t];  // 256 threads x 32B = 8KB
      const int d = t >> 1, half = t & 1;
      *(v16h*)&Vl[d][half * 16] =
          *(const v16h*)(Vb + (size_t)d * NK + key0 + half * 16);
      if (kt + 1 < NK / 32)
        __builtin_prefetch(Kb + (size_t)(key0 + 32) * DIM + t * 16, 0, 1);
    }
    __syncthreads();  // staging visible to all waves

    // ---- S = Q @ K^T for 32 keys (two 16x16 C-frags), K-dim in 4 chunks ----
    v8f s0, s1;
    #pragma unroll
    for (int i = 0; i < 8; ++i) { s0[i] = 0.0f; s1[i] = 0.0f; }
    #pragma unroll
    for (int c = 0; c < 4; ++c) {
      // B-frag lane data = Kl[key n][c*32 + 16*hl .. +16) : contiguous 32B
      v16h b0 = *(const v16h*)&Kl[nlo][c * 32 + hl * 16];
      v16h b1 = *(const v16h*)&Kl[16 + nlo][c * 32 + hl * 16];
      s0 = wmma_f16(qf[c], b0, s0);
      s1 = wmma_f16(qf[c], b1, s1);
    }

    const float bias0 = kbb[key0 + nlo];
    const float bias1 = kbb[key0 + 16 + nlo];

    // ---- online softmax (rows live across the 16-lane half-groups) ----
    float p0[8], p1[8], alpha[8];
    #pragma unroll
    for (int v = 0; v < 8; ++v) {
      float x0 = s0[v] * SCALE + bias0;
      float x1 = s1[v] * SCALE + bias1;
      float mx = fmaxf(x0, x1);
      #pragma unroll
      for (int msk = 1; msk < 16; msk <<= 1)
        mx = fmaxf(mx, __shfl_xor(mx, msk, 32));
      float nm = fmaxf(rm[v], mx);
      float al = __expf(rm[v] - nm);
      float e0 = __expf(x0 - nm);
      float e1 = __expf(x1 - nm);
      float ts = e0 + e1;
      #pragma unroll
      for (int msk = 1; msk < 16; msk <<= 1)
        ts += __shfl_xor(ts, msk, 32);
      rs[v] = rs[v] * al + ts;
      rm[v] = nm;
      alpha[v] = al; p0[v] = e0; p1[v] = e1;
    }
    #pragma unroll
    for (int c = 0; c < 8; ++c)
      #pragma unroll
      for (int v = 0; v < 8; ++v) of[c][v] *= alpha[v];

    // ---- C-layout P -> LDS -> A-layout fragment (same-wave DS in-order) ----
    #pragma unroll
    for (int v = 0; v < 8; ++v) {
      const int m = v + 8 * hl;
      pbuf[wid][m][nlo]      = (_Float16)p0[v];
      pbuf[wid][m][nlo + 16] = (_Float16)p1[v];
    }
    asm volatile("s_wait_dscnt 0" ::: "memory");
    v16h pa;
    #pragma unroll
    for (int v = 0; v < 8; ++v) {
      int k0 = a_koff(v, hl);
      pa[2 * v]     = pbuf[wid][nlo][k0];
      pa[2 * v + 1] = pbuf[wid][nlo][k0 + 1];
    }

    // ---- O += P(16x32) @ V(32x128): 8 WMMAs over d-chunks ----
    #pragma unroll
    for (int c = 0; c < 8; ++c) {
      // B-frag lane data = Vl[c*16+n][16*hl .. +16) : contiguous 32B
      v16h vb = *(const v16h*)&Vl[c * 16 + nlo][hl * 16];
      of[c] = wmma_f16(pa, vb, of[c]);
    }
  }

  // ---- normalize and write O (f32) ----
  float inv[8];
  #pragma unroll
  for (int v = 0; v < 8; ++v) inv[v] = 1.0f / rs[v];
  float* Ob = O + ((size_t)b * NQ + qrow0) * DIM;
  #pragma unroll
  for (int c = 0; c < 8; ++c)
    #pragma unroll
    for (int v = 0; v < 8; ++v)
      Ob[(size_t)(v + 8 * hl) * DIM + c * 16 + nlo] = of[c][v] * inv[v];
}

// ---------------------------------------------------------------------------
extern "C" void kernel_launch(void* const* d_in, const int* in_sizes, int n_in,
                              void* d_out, int out_size, void* d_ws,
                              size_t ws_size, hipStream_t stream) {
  (void)in_sizes; (void)n_in; (void)out_size; (void)ws_size;
  const float* query    = (const float*)d_in[0];
  const float* key      = (const float*)d_in[1];
  const float* value    = (const float*)d_in[2];
  const float* att_bias = (const float*)d_in[3];
  const float* q_w = (const float*)d_in[4];
  const float* q_b = (const float*)d_in[5];
  const float* k_w = (const float*)d_in[6];
  const float* v_w = (const float*)d_in[7];
  const float* v_b = (const float*)d_in[8];
  const float* vs_w = (const float*)d_in[9];
  const float* vs_b = (const float*)d_in[10];
  const float* p_w = (const float*)d_in[11];
  const float* p_b = (const float*)d_in[12];

  // Workspace: Qp(4MB) Kp(4MB) Vt(4MB) kb(64KB) O(8MB)  ~20MB total
  _Float16* Qp = (_Float16*)d_ws;
  _Float16* Kp = Qp + (size_t)BATCH * NQ * DIM;
  _Float16* Vt = Kp + (size_t)BATCH * NK * DIM;
  float* kbias = (float*)(Vt + (size_t)BATCH * NK * DIM);
  float* Obuf  = kbias + (size_t)BATCH * NK;

  const int rq = BATCH * NQ, rk = BATCH * NK;
  dim3 blk(256);

  gemm128_wmma<_Float16, false><<<rq / 128, blk, 0, stream>>>(query, q_w, q_b, Qp, rq, 1);
  gemm128_wmma<_Float16, false><<<rk / 128, blk, 0, stream>>>(key, k_w, nullptr, Kp, rk, 1);
  gemm128_wmma<_Float16, true ><<<rk / 128, blk, 0, stream>>>(value, v_w, v_b, Vt, rk, NK);
  kbias_kernel<<<(rk + 255) / 256, blk, 0, stream>>>(value, att_bias, vs_w, vs_b, kbias);
  flash_attn_wmma<<<rq / 128, blk, 0, stream>>>(Qp, Kp, Vt, kbias, Obuf);
  gemm128_wmma<float, false><<<rq / 128, blk, 0, stream>>>(Obuf, p_w, p_b, (float*)d_out, rq, 1);
}
